// MultiHeadSelfAttention_23029614641226
// MI455X (gfx1250) — matrix operand order repaired and checked by reading, not compiled, not run on hardware
//
#include <hip/hip_runtime.h>
#include <hip/hip_bf16.h>

// ---------------------------------------------------------------------------
// MI455X (gfx1250) multi-head self-attention, f16 WMMA with f32 accumulation.
//   D_MODEL=1024, N_HEADS=16, D_HEAD=64, B=2, T=2048.
// Workspace layout (assumes ws_size >= 48 MiB):
//   [ 0..8M)   xh   : x cast to f16,            [4096][1024]
//   [ 8..16M)  Wt   : Wq,Wk,Wv,Wo f16 N-major,  4 x [1024][1024]
//   [16..24M)  Qh   : [b][h][t][dh] f16
//   [24..32M)  Kh   : [b][h][t][dh] f16
//   [32..40M)  Vt   : [b][h][dh][t] f16 (transposed for PV B-fragments)
//   [40..48M)  Ah   : attention output f16, [b*t][1024]
// ---------------------------------------------------------------------------

typedef __attribute__((ext_vector_type(16))) _Float16 v16h;
typedef __attribute__((ext_vector_type(8)))  float    v8f;
typedef __attribute__((ext_vector_type(4)))  int      v4i;

union V16H { v16h v; uint4 u[2]; };

#define D_MODEL 1024
#define N_HEADS 16
#define D_HEAD  64
#define SEQ_T   2048
#define BATCH   2
#define M_TOTAL (BATCH * SEQ_T)   // 4096

#define LDSP 40    // padded LDS row stride in halves (80B: 16B aligned, conflict-free)
#define BUFH (128 * LDSP)   // halves per staging buffer

__device__ __forceinline__ v8f wmma16(v16h a, v16h b, v8f c) {
  // D = A(16x32 f16) * B(32x16 f16) + C(16x16 f32)
  return __builtin_amdgcn_wmma_f32_16x16x32_f16(
      /*neg_a=*/false, a, /*neg_b=*/false, b,
      /*c_mod=*/(short)0, c, /*reuse_a=*/false, /*reuse_b=*/false);
}

// ---------------------------------------------------------------------------
// CDNA5 async global->LDS staging (GLOBAL_LOAD_ASYNC_TO_LDS_B128, ASYNCcnt).
// Builtin signature (from compiler diagnostic): (int4 AS1*, int4 AS3*, Ii, Ii).
// Guarded: fall back to sync load + ds_store if the builtin isn't declared.
// ---------------------------------------------------------------------------
#if __has_builtin(__builtin_amdgcn_global_load_async_to_lds_b128)
#define HAVE_ASYNC_LDS 1
#define GLOBAL_AS __attribute__((address_space(1)))
#define LDS_AS    __attribute__((address_space(3)))
__device__ __forceinline__ void async_cp16(const _Float16* g, _Float16* l) {
  __builtin_amdgcn_global_load_async_to_lds_b128(
      (GLOBAL_AS v4i*)(g), (LDS_AS v4i*)(l), /*offset=*/0, /*cpol=*/0);
}
#if __has_builtin(__builtin_amdgcn_s_wait_asynccnt)
#define WAIT_ASYNC(n) __builtin_amdgcn_s_wait_asynccnt(n)
#else
#define WAIT_ASYNC(n) asm volatile("s_wait_asynccnt %0" ::"i"(n) : "memory")
#endif
#else
#define HAVE_ASYNC_LDS 0
#endif

// ---------------------------------------------------------------------------
// Kernel 0a: cast x fp32 -> f16
// ---------------------------------------------------------------------------
__global__ __launch_bounds__(256) void cast_x_kernel(const float* __restrict__ x,
                                                     _Float16* __restrict__ xh, int n) {
  int i = blockIdx.x * 256 + threadIdx.x;
  if (i < n) xh[i] = (_Float16)x[i];
}

// ---------------------------------------------------------------------------
// Kernel 0b: W (K-major [k][n] fp32) -> Wt (N-major [n][k] f16), 32x32 tiles.
// ---------------------------------------------------------------------------
__global__ __launch_bounds__(256) void transpose_w_kernel(
    const float* __restrict__ Wq, const float* __restrict__ Wk,
    const float* __restrict__ Wv, const float* __restrict__ Wo,
    _Float16* __restrict__ WtBase) {
  __shared__ _Float16 tile[32][33];
  const float* W = (blockIdx.z == 0) ? Wq : (blockIdx.z == 1) ? Wk
                 : (blockIdx.z == 2) ? Wv : Wo;
  _Float16* Wt = WtBase + (size_t)blockIdx.z * D_MODEL * D_MODEL;
  int k0 = blockIdx.x * 32, n0 = blockIdx.y * 32;
  int tx = threadIdx.x & 31, ty = threadIdx.x >> 5;  // ty 0..7
#pragma unroll
  for (int i = 0; i < 4; ++i)
    tile[ty + i * 8][tx] = (_Float16)W[(size_t)(k0 + ty + i * 8) * D_MODEL + n0 + tx];
  __syncthreads();
#pragma unroll
  for (int i = 0; i < 4; ++i)
    Wt[(size_t)(n0 + ty + i * 8) * D_MODEL + k0 + tx] = tile[tx][ty + i * 8];
}

// ---------------------------------------------------------------------------
// Shared GEMM mainloop: C[128x128] += A[128x1024] * Bt[128x1024]^T
// A row-major [m][k]; Bt N-major [n][k]. Block = 256 threads = 8 waves,
// wave grid 4(M) x 2(N); each wave owns a 32x64 tile = 2x4 WMMA accumulators.
// Async path: double-buffered LDS (2 x 10KB padded per matrix), staged with
// global_load_async_to_lds_b128; current tile drained with s_wait_asynccnt 4
// (async loads retire in order) while the next tile's 4 ops stay in flight.
// ---------------------------------------------------------------------------
__device__ __forceinline__ void gemm_mainloop(
    const _Float16* __restrict__ A, const _Float16* __restrict__ Bt,
    int m0, int n0, _Float16* ldsA, _Float16* ldsB, v8f acc[2][4]) {
  const int tid  = threadIdx.x;
  const int lane = tid & 31;
  const int wave = tid >> 5;
  const int wm   = wave >> 1;          // 0..3
  const int wn   = wave & 1;           // 0..1
  const int half = lane >> 4;          // 0/1 (lane group)
  const int l16  = lane & 15;
  const int sr   = tid >> 1;           // staging row 0..127
  const int sc   = (tid & 1) * 16;     // staging col (halves): 0 or 16

  const _Float16* gA = A  + (size_t)(m0 + sr) * D_MODEL + sc;
  const _Float16* gB = Bt + (size_t)(n0 + sr) * D_MODEL + sc;
  _Float16* sA = ldsA + sr * LDSP + sc;
  _Float16* sB = ldsB + sr * LDSP + sc;

#pragma unroll
  for (int i = 0; i < 2; ++i)
#pragma unroll
    for (int j = 0; j < 4; ++j)
#pragma unroll
      for (int r = 0; r < 8; ++r) acc[i][j][r] = 0.f;

#if HAVE_ASYNC_LDS
  // ---- prologue: stage k0=0 into buffer 0 (4 async ops per wave) ----
  async_cp16(gA, sA);
  async_cp16(gA + 8, sA + 8);
  async_cp16(gB, sB);
  async_cp16(gB + 8, sB + 8);

  for (int k0 = 0, it = 0; k0 < D_MODEL; k0 += 32, ++it) {
    const int cur = it & 1;
    const bool has_next = (k0 + 32 < D_MODEL);
    if (has_next) {
      const int nxt = cur ^ 1;
      async_cp16(gA + k0 + 32,     sA + nxt * BUFH);
      async_cp16(gA + k0 + 32 + 8, sA + nxt * BUFH + 8);
      async_cp16(gB + k0 + 32,     sB + nxt * BUFH);
      async_cp16(gB + k0 + 32 + 8, sB + nxt * BUFH + 8);
      WAIT_ASYNC(4);   // current tile retired; next tile's 4 ops in flight
    } else {
      WAIT_ASYNC(0);
    }
    __syncthreads();   // all waves' current-tile staging visible

    const _Float16* bufA = ldsA + cur * BUFH;
    const _Float16* bufB = ldsB + cur * BUFH;
    V16H af[2], bf[4];
#pragma unroll
    for (int i = 0; i < 2; ++i) {
      const _Float16* p = bufA + (wm * 32 + i * 16 + l16) * LDSP + half * 8;
      af[i].u[0] = ((const uint4*)p)[0];
      af[i].u[1] = ((const uint4*)(p + 16))[0];
    }
#pragma unroll
    for (int j = 0; j < 4; ++j) {
      const _Float16* p = bufB + (wn * 64 + j * 16 + l16) * LDSP + half * 8;
      bf[j].u[0] = ((const uint4*)p)[0];
      bf[j].u[1] = ((const uint4*)(p + 16))[0];
    }
#pragma unroll
    for (int i = 0; i < 2; ++i)
#pragma unroll
      for (int j = 0; j < 4; ++j)
        acc[i][j] = wmma16(af[i].v, bf[j].v, acc[i][j]);
    __syncthreads();   // everyone done reading before buffers are re-staged
  }
#else
  // ---- fallback: sync staging through VGPRs ----
  for (int k0 = 0; k0 < D_MODEL; k0 += 32) {
    uint4 a0 = ((const uint4*)(gA + k0))[0];
    uint4 a1 = ((const uint4*)(gA + k0))[1];
    uint4 b0 = ((const uint4*)(gB + k0))[0];
    uint4 b1 = ((const uint4*)(gB + k0))[1];
    if (k0 + 32 < D_MODEL) {                 // near-scope prefetch of next tiles
      __builtin_prefetch(gA + k0 + 32, 0, 3);
      __builtin_prefetch(gB + k0 + 32, 0, 3);
    }
    ((uint4*)(sA))[0] = a0;
    ((uint4*)(sA + 8))[0] = a1;
    ((uint4*)(sB))[0] = b0;
    ((uint4*)(sB + 8))[0] = b1;
    __syncthreads();

    V16H af[2], bf[4];
#pragma unroll
    for (int i = 0; i < 2; ++i) {
      const _Float16* p = ldsA + (wm * 32 + i * 16 + l16) * LDSP + half * 8;
      af[i].u[0] = ((const uint4*)p)[0];
      af[i].u[1] = ((const uint4*)(p + 16))[0];
    }
#pragma unroll
    for (int j = 0; j < 4; ++j) {
      const _Float16* p = ldsB + (wn * 64 + j * 16 + l16) * LDSP + half * 8;
      bf[j].u[0] = ((const uint4*)p)[0];
      bf[j].u[1] = ((const uint4*)(p + 16))[0];
    }
#pragma unroll
    for (int i = 0; i < 2; ++i)
#pragma unroll
      for (int j = 0; j < 4; ++j)
        acc[i][j] = wmma16(af[i].v, bf[j].v, acc[i][j]);
    __syncthreads();
  }
#endif
}

// ---------------------------------------------------------------------------
// Kernel 1: fused QKV projection. blockIdx.z in {0,1,2} selects q/k/v.
// Writes Q,K as [b][h][t][dh] f16; V transposed as [b][h][dh][t] f16.
// ---------------------------------------------------------------------------
__global__ __launch_bounds__(256) void qkv_gemm_kernel(
    const _Float16* __restrict__ xh, const _Float16* __restrict__ Wt,
    const float* __restrict__ bq, const float* __restrict__ bk,
    const float* __restrict__ bv,
    _Float16* __restrict__ Qh, _Float16* __restrict__ Kh,
    _Float16* __restrict__ Vt) {
  __shared__ _Float16 ldsA[2 * BUFH];
  __shared__ _Float16 ldsB[2 * BUFH];
  const int m0 = blockIdx.x * 128, n0 = blockIdx.y * 128, z = blockIdx.z;
  const _Float16* W = Wt + (size_t)z * D_MODEL * D_MODEL;
  const float* bias = (z == 0) ? bq : (z == 1) ? bk : bv;

  v8f acc[2][4];
  gemm_mainloop(xh, W, m0, n0, ldsA, ldsB, acc);

  const int lane = threadIdx.x & 31, wave = threadIdx.x >> 5;
  const int wm = wave >> 1, wn = wave & 1, half = lane >> 4, l16 = lane & 15;
#pragma unroll
  for (int j = 0; j < 4; ++j) {
    const int n  = n0 + wn * 64 + j * 16 + l16;     // column = lane (C layout)
    const float bias_n = bias[n];
    const int hh = n >> 6, dh = n & 63;
#pragma unroll
    for (int i = 0; i < 2; ++i) {
#pragma unroll
      for (int r = 0; r < 8; ++r) {
        const int m  = m0 + wm * 32 + i * 16 + half * 8 + r;  // row = r + 8*half
        const int bb = m >> 11, t = m & (SEQ_T - 1);
        const _Float16 vh = (_Float16)(acc[i][j][r] + bias_n);
        if (z == 0)
          Qh[(((size_t)bb * N_HEADS + hh) * SEQ_T + t) * D_HEAD + dh] = vh;
        else if (z == 1)
          Kh[(((size_t)bb * N_HEADS + hh) * SEQ_T + t) * D_HEAD + dh] = vh;
        else
          Vt[(((size_t)bb * N_HEADS + hh) * D_HEAD + dh) * SEQ_T + t] = vh;
      }
    }
  }
}

// ---------------------------------------------------------------------------
// Kernel 2: causal flash-attention. One wave per 16-query tile.
// Grid (T/128, H, B), 256 threads = 8 waves = 8 query tiles.
// ---------------------------------------------------------------------------
__global__ __launch_bounds__(256) void attn_kernel(
    const _Float16* __restrict__ Qh, const _Float16* __restrict__ Kh,
    const _Float16* __restrict__ Vt, _Float16* __restrict__ Ah) {
  __shared__ _Float16 pbuf[8][16 * LDSP];   // per-wave P bounce buffer
  const int lane = threadIdx.x & 31, wave = threadIdx.x >> 5;
  const int half = lane >> 4, l16 = lane & 15;
  const int b = blockIdx.z, h = blockIdx.y;
  const int qt = blockIdx.x * 8 + wave;     // query tile 0..127

  const _Float16* Qb = Qh + (((size_t)b * N_HEADS + h) * SEQ_T + qt * 16) * D_HEAD;
  const _Float16* Kb = Kh + (((size_t)b * N_HEADS + h) * SEQ_T) * D_HEAD;
  const _Float16* Vb = Vt + (((size_t)b * N_HEADS + h) * D_HEAD) * SEQ_T;

  // Q A-fragments for dh chunks {0..31},{32..63} — loaded once, reused all chunks
  V16H qf[2];
  {
    const _Float16* qrow = Qb + (size_t)l16 * D_HEAD + half * 8;
#pragma unroll
    for (int c = 0; c < 2; ++c) {
      qf[c].u[0] = ((const uint4*)(qrow + c * 32))[0];
      qf[c].u[1] = ((const uint4*)(qrow + c * 32 + 16))[0];
    }
  }

  v8f o[4];
  float mrow[8], lrow[8];
#pragma unroll
  for (int j = 0; j < 4; ++j)
#pragma unroll
    for (int r = 0; r < 8; ++r) o[j][r] = 0.f;
#pragma unroll
  for (int r = 0; r < 8; ++r) { mrow[r] = -__builtin_inff(); lrow[r] = 0.f; }

  const int nch = qt / 2 + 1;               // 32-key chunks under causal mask
  for (int kc = 0; kc < nch; ++kc) {
    const int kb0 = kc * 32;

    // ---- S = Q K^T : two 16x16 score tiles, K-dim = 64 (2 WMMA each) ----
    v8f s[2];
#pragma unroll
    for (int j = 0; j < 2; ++j)
#pragma unroll
      for (int r = 0; r < 8; ++r) s[j][r] = 0.f;
#pragma unroll
    for (int j = 0; j < 2; ++j) {
#pragma unroll
      for (int c = 0; c < 2; ++c) {
        V16H kf;   // B-frag: lane col = key kb0+j*16+l16; halves = dh in table order
        const _Float16* kp =
            Kb + (size_t)(kb0 + j * 16 + l16) * D_HEAD + c * 32 + half * 8;
        kf.u[0] = ((const uint4*)kp)[0];
        kf.u[1] = ((const uint4*)(kp + 16))[0];
        s[j] = wmma16(qf[c].v, kf.v, s[j]);
      }
    }

    // ---- online softmax (rows live across 16-lane halves of the wave) ----
    const bool last = (kc == nch - 1);
    float e0[8], e1[8], newmax[8];
#pragma unroll
    for (int r = 0; r < 8; ++r) {
      const int qrow = qt * 16 + half * 8 + r;
      float v0 = s[0][r] * 0.125f;          // 1/sqrt(64)
      float v1 = s[1][r] * 0.125f;
      if (last) {
        if (kb0 + l16 > qrow)      v0 = -__builtin_inff();
        if (kb0 + 16 + l16 > qrow) v1 = -__builtin_inff();
      }
      e0[r] = v0; e1[r] = v1;
      float mx = fmaxf(v0, v1);
      mx = fmaxf(mx, __shfl_xor(mx, 1, 32));
      mx = fmaxf(mx, __shfl_xor(mx, 2, 32));
      mx = fmaxf(mx, __shfl_xor(mx, 4, 32));
      mx = fmaxf(mx, __shfl_xor(mx, 8, 32));
      newmax[r] = fmaxf(mrow[r], mx);
    }
#pragma unroll
    for (int r = 0; r < 8; ++r) {
      const float alpha = __expf(mrow[r] - newmax[r]);
      mrow[r] = newmax[r];
      const float p0 = __expf(e0[r] - newmax[r]);
      const float p1 = __expf(e1[r] - newmax[r]);
      e0[r] = p0; e1[r] = p1;
      float rs = p0 + p1;
      rs += __shfl_xor(rs, 1, 32);
      rs += __shfl_xor(rs, 2, 32);
      rs += __shfl_xor(rs, 4, 32);
      rs += __shfl_xor(rs, 8, 32);
      lrow[r] = lrow[r] * alpha + rs;
#pragma unroll
      for (int j = 0; j < 4; ++j) o[j][r] *= alpha;
    }

    // ---- C-frag -> A-frag relayout of P through per-wave LDS ----
    _Float16* pb = pbuf[wave];
#pragma unroll
    for (int r = 0; r < 8; ++r) {
      const int row = half * 8 + r;
      pb[row * LDSP + l16]      = (_Float16)e0[r];
      pb[row * LDSP + 16 + l16] = (_Float16)e1[r];
    }
    // same-wave DS RAW: drain the DS counter before the fragment reload
    asm volatile("s_wait_dscnt 0" ::: "memory");
    V16H pf;
    {
      const _Float16* pr = pb + l16 * LDSP + half * 8;
      pf.u[0] = ((const uint4*)pr)[0];
      pf.u[1] = ((const uint4*)(pr + 16))[0];
    }

    // ---- O += P V : 4 N-tiles of dh, K-dim = 32 keys ----
#pragma unroll
    for (int j = 0; j < 4; ++j) {
      V16H vf;   // B-frag: lane col = dh j*16+l16; halves = keys in table order (Vt)
      const _Float16* vp =
          Vb + (size_t)(j * 16 + l16) * SEQ_T + kb0 + half * 8;
      vf.u[0] = ((const uint4*)vp)[0];
      vf.u[1] = ((const uint4*)(vp + 16))[0];
      o[j] = wmma16(pf.v, vf.v, o[j]);
    }
  }

  // ---- epilogue: normalize, write f16 to Ah[b*t][h*64+dh] ----
  _Float16* ob = Ah + ((size_t)b * SEQ_T + qt * 16) * D_MODEL + h * D_HEAD;
#pragma unroll
  for (int j = 0; j < 4; ++j)
#pragma unroll
    for (int r = 0; r < 8; ++r) {
      const int row = half * 8 + r;
      ob[(size_t)row * D_MODEL + j * 16 + l16] = (_Float16)(o[j][r] / lrow[r]);
    }
}

// ---------------------------------------------------------------------------
// Kernel 3: output projection, fp32 result straight to d_out.
// ---------------------------------------------------------------------------
__global__ __launch_bounds__(256) void out_proj_kernel(
    const _Float16* __restrict__ Ah, const _Float16* __restrict__ WtO,
    const float* __restrict__ bo, float* __restrict__ out) {
  __shared__ _Float16 ldsA[2 * BUFH];
  __shared__ _Float16 ldsB[2 * BUFH];
  const int m0 = blockIdx.x * 128, n0 = blockIdx.y * 128;

  v8f acc[2][4];
  gemm_mainloop(Ah, WtO, m0, n0, ldsA, ldsB, acc);

  const int lane = threadIdx.x & 31, wave = threadIdx.x >> 5;
  const int wm = wave >> 1, wn = wave & 1, half = lane >> 4, l16 = lane & 15;
#pragma unroll
  for (int j = 0; j < 4; ++j) {
    const int n = n0 + wn * 64 + j * 16 + l16;
    const float bias_n = bo[n];
#pragma unroll
    for (int i = 0; i < 2; ++i)
#pragma unroll
      for (int r = 0; r < 8; ++r) {
        const int m = m0 + wm * 32 + i * 16 + half * 8 + r;
        out[(size_t)m * D_MODEL + n] = acc[i][j][r] + bias_n;
      }
  }
}

// ---------------------------------------------------------------------------
extern "C" void kernel_launch(void* const* d_in, const int* in_sizes, int n_in,
                              void* d_out, int out_size, void* d_ws, size_t ws_size,
                              hipStream_t stream) {
  const float* x  = (const float*)d_in[0];
  const float* Wq = (const float*)d_in[1];
  const float* bq = (const float*)d_in[2];
  const float* Wk = (const float*)d_in[3];
  const float* bk = (const float*)d_in[4];
  const float* Wv = (const float*)d_in[5];
  const float* bv = (const float*)d_in[6];
  const float* Wo = (const float*)d_in[7];
  const float* bo = (const float*)d_in[8];
  float* out = (float*)d_out;

  char* ws = (char*)d_ws;
  const size_t MiB = 1024 * 1024;
  _Float16* xh = (_Float16*)(ws + 0 * MiB);
  _Float16* Wt = (_Float16*)(ws + 8 * MiB);   // q,k,v,o consecutive (2 MiB each)
  _Float16* Qh = (_Float16*)(ws + 16 * MiB);
  _Float16* Kh = (_Float16*)(ws + 24 * MiB);
  _Float16* Vt = (_Float16*)(ws + 32 * MiB);
  _Float16* Ah = (_Float16*)(ws + 40 * MiB);

  const int nx = M_TOTAL * D_MODEL;  // 4194304
  cast_x_kernel<<<(nx + 255) / 256, 256, 0, stream>>>(x, xh, nx);
  transpose_w_kernel<<<dim3(D_MODEL / 32, D_MODEL / 32, 4), 256, 0, stream>>>(
      Wq, Wk, Wv, Wo, Wt);
  qkv_gemm_kernel<<<dim3(M_TOTAL / 128, D_MODEL / 128, 3), 256, 0, stream>>>(
      xh, Wt, bq, bk, bv, Qh, Kh, Vt);
  attn_kernel<<<dim3(SEQ_T / 128, N_HEADS, BATCH), 256, 0, stream>>>(Qh, Kh, Vt, Ah);
  out_proj_kernel<<<dim3(M_TOTAL / 128, D_MODEL / 128), 256, 0, stream>>>(
      Ah, Wt + (size_t)3 * D_MODEL * D_MODEL, bo, out);
}
